// SpatialShiftCorrAggr_31404800868807
// MI455X (gfx1250) — matrix operand order repaired and checked by reading, not compiled, not run on hardware
//
#include <hip/hip_runtime.h>
#include <hip/hip_bf16.h>

typedef __attribute__((ext_vector_type(16))) _Float16 v16h;
typedef __attribute__((ext_vector_type(8)))  _Float16 h8;
typedef __attribute__((ext_vector_type(8)))  float    v8f;

#define B_   4
#define C_   64
#define H_   96
#define W_   96
#define HW_  (H_*W_)          // 9216
#define NPIX (B_*HW_)         // 36864

// ---------------------------------------------------------------------------
// WMMA operand loaders (CDNA5 v_wmma_f32_16x16x32_f16 VGPR layouts, ISA 7.12.2)
// ---------------------------------------------------------------------------
__device__ __forceinline__ v16h cat8(h8 lo, h8 hi) {
  return __builtin_shufflevector(lo, hi, 0,1,2,3,4,5,6,7,8,9,10,11,12,13,14,15);
}
// A (16x32 MxK): lane = 16*sel + m holds row m.
// v16h elems 0..7  <- K[kb + 8*sel  .. +7]   (VGPR0-3)
// v16h elems 8..15 <- K[kb + 16 + 8*sel .. +7] (VGPR4-7)
__device__ __forceinline__ v16h load_a32(const _Float16* row_kb, int sel) {
  h8 lo = *(const h8*)(row_kb + 8 * sel);
  h8 hi = *(const h8*)(row_kb + 16 + 8 * sel);
  return cat8(lo, hi);
}
// B (32x16 KxN): lane = 16*sel + n holds column n; elems 0..7 <- K[kb+8*sel .. +7], elems 8..15 <- K[kb+16+8*sel .. +7]
// requires weights stored transposed: Wt[n][k] contiguous in k.
__device__ __forceinline__ v16h load_b32(const _Float16* row_kb, int sel) {
  h8 lo = *(const h8*)(row_kb + 8 * sel);
  h8 hi = *(const h8*)(row_kb + 16 + 8 * sel);
  return cat8(lo, hi);
}

// ---------------------------------------------------------------------------
// Stage 0: layout prep.  NCHW f32 -> f16 NHWC copies:
//   fused[pix][0:64]  = sup_frame   (channels 64..255 filled by natten later)
//   x128 [pix][0:64]  = sup_frame   (channels 64..127 filled by smooth conv)
//   qh   [pix][0:64]  = key_frame
// ---------------------------------------------------------------------------
__global__ void prep_kernel(const float* __restrict__ sup, const float* __restrict__ key,
                            _Float16* __restrict__ fused, _Float16* __restrict__ x128,
                            _Float16* __restrict__ qh) {
  int id = blockIdx.x * blockDim.x + threadIdx.x;   // pid*64 + c
  if (id >= NPIX * C_) return;
  int c = id & 63;
  int pid = id >> 6;
  int x = pid % W_, y = (pid / W_) % H_, b = pid / HW_;
  int nchw = ((b * C_ + c) * H_ + y) * W_ + x;
  _Float16 s = (_Float16)sup[nchw];
  fused[pid * 256 + c] = s;
  x128[pid * 128 + c]  = s;
  qh[pid * 64 + c]     = (_Float16)key[nchw];
}

// ---------------------------------------------------------------------------
// Stage 1: group-shift + grouped 3x3 conv (groups=8), fused.
// shifted[c][y][x] = sup[c][y-sh][x-sw] (zero OOB), then 3x3 conv pad=1.
// Combined tap: sup[g*8+icl][y+ky-1-sh][x+kx-1-sw], zero OOB.
// Result -> x128[pix][64+oc] as f16.
// ---------------------------------------------------------------------------
__global__ void smooth_kernel(const float* __restrict__ sup, const float* __restrict__ w,
                              const float* __restrict__ bias, _Float16* __restrict__ x128) {
  // shifts = [(sh,sw) for sh in (-3,0,3) for sw in (-3,0,3) if not both 0]
  const int SHY[8] = {-3,-3,-3, 0, 0, 3, 3, 3};
  const int SHX[8] = {-3, 0, 3,-3, 3,-3, 0, 3};
  int id = blockIdx.x * blockDim.x + threadIdx.x;   // pid*64 + oc
  if (id >= NPIX * C_) return;
  int oc = id & 63;
  int pid = id >> 6;
  int x = pid % W_, y = (pid / W_) % H_, b = pid / HW_;
  int g = oc >> 3;
  int sh = SHY[g], sw = SHX[g];
  float acc = bias[oc];
  #pragma unroll
  for (int icl = 0; icl < 8; ++icl) {
    const float* sp = sup + ((size_t)(b * C_ + g * 8 + icl)) * HW_;
    #pragma unroll
    for (int ky = 0; ky < 3; ++ky) {
      int yy = y + ky - 1 - sh;
      if (yy < 0 || yy >= H_) continue;
      #pragma unroll
      for (int kx = 0; kx < 3; ++kx) {
        int xx = x + kx - 1 - sw;
        if (xx < 0 || xx >= W_) continue;
        acc += w[((oc * 8 + icl) * 3 + ky) * 3 + kx] * sp[yy * W_ + xx];
      }
    }
  }
  x128[pid * 128 + 64 + oc] = (_Float16)acc;
}

// ---------------------------------------------------------------------------
// Weight conversion helpers
// ---------------------------------------------------------------------------
__global__ void cast_f2h(const float* __restrict__ src, _Float16* __restrict__ dst, int n) {
  int id = blockIdx.x * blockDim.x + threadIdx.x;
  if (id < n) dst[id] = (_Float16)src[id];
}
// fusion_w OIHW (64,256,3,3) -> wft[(ky*3+kx)][oc][ic] f16
__global__ void wt_fusion_kernel(const float* __restrict__ w, _Float16* __restrict__ wt) {
  int id = blockIdx.x * blockDim.x + threadIdx.x;
  if (id >= 64 * 256 * 9) return;
  int kx = id % 3;
  int ky = (id / 3) % 3;
  int ic = (id / 9) % 256;
  int oc = id / (9 * 256);
  wt[(((ky * 3 + kx) * 64 + oc) * 256) + ic] = (_Float16)w[id];
}

// ---------------------------------------------------------------------------
// Stage 2: conv_reduce 1x1: sup2[pix][64] = x128[pix][128] * Wr^T  (WMMA GEMM)
// Block = 4 waves; wave w -> n-tile w; block -> 16-pixel m-tile.
// ---------------------------------------------------------------------------
__global__ void gemm_reduce_kernel(const _Float16* __restrict__ X,   // [NPIX][128]
                                   const _Float16* __restrict__ Wt,  // [64][128]
                                   const float* __restrict__ bias,
                                   _Float16* __restrict__ Y) {       // [NPIX][64]
  int lane = threadIdx.x & 31;
  int wave = threadIdx.x >> 5;
  int m0 = blockIdx.x * 16;
  int n0 = wave * 16;
  int sel = lane >> 4;
  int mr = lane & 15;
  int nc = lane & 15;
  v8f c = {};
  const _Float16* arow = X + (size_t)(m0 + mr) * 128;
  const _Float16* brow = Wt + (size_t)(n0 + nc) * 128;
  #pragma unroll
  for (int kb = 0; kb < 128; kb += 32) {
    v16h a = load_a32(arow + kb, sel);
    v16h bm = load_b32(brow + kb, sel);
    c = __builtin_amdgcn_wmma_f32_16x16x32_f16(false, a, false, bm, (short)0, c, false, false);
  }
  float bz = bias[n0 + nc];
  #pragma unroll
  for (int r = 0; r < 8; ++r) {
    int m = m0 + r + 8 * sel;
    Y[(size_t)m * 64 + n0 + nc] = (_Float16)(c[r] + bz);
  }
}

// ---------------------------------------------------------------------------
// Stage 3: three v 1x1 convs: vh[lvl][pix][64] = sup2 * Wv[lvl]^T  (WMMA)
// grid = (2304 m-tiles, 3 levels)
// ---------------------------------------------------------------------------
__global__ void gemm_v_kernel(const _Float16* __restrict__ X,    // [NPIX][64]
                              const _Float16* __restrict__ Wt,   // [3][64][64]
                              const float* __restrict__ b0, const float* __restrict__ b1,
                              const float* __restrict__ b2,
                              _Float16* __restrict__ Y) {        // [3][NPIX][64]
  int lvl = blockIdx.y;
  const _Float16* wt = Wt + (size_t)lvl * 64 * 64;
  const float* bias = (lvl == 0) ? b0 : ((lvl == 1) ? b1 : b2);
  _Float16* y = Y + (size_t)lvl * NPIX * 64;
  int lane = threadIdx.x & 31;
  int wave = threadIdx.x >> 5;
  int m0 = blockIdx.x * 16;
  int n0 = wave * 16;
  int sel = lane >> 4;
  int mr = lane & 15;
  int nc = lane & 15;
  v8f c = {};
  const _Float16* arow = X + (size_t)(m0 + mr) * 64;
  const _Float16* brow = wt + (size_t)(n0 + nc) * 64;
  #pragma unroll
  for (int kb = 0; kb < 64; kb += 32) {
    v16h a = load_a32(arow + kb, sel);
    v16h bm = load_b32(brow + kb, sel);
    c = __builtin_amdgcn_wmma_f32_16x16x32_f16(false, a, false, bm, (short)0, c, false, false);
  }
  float bz = bias[n0 + nc];
  #pragma unroll
  for (int r = 0; r < 8; ++r) {
    int m = m0 + r + 8 * sel;
    y[(size_t)m * 64 + n0 + nc] = (_Float16)(c[r] + bz);
  }
}

// ---------------------------------------------------------------------------
// Stage 4: NATTEN level (clipped windows + RPB + softmax), one thread = pixel.
// q = key_frame (qh), k = sup2, v = vh[lvl]; result -> fused[pix][ch_off..+63]
// ---------------------------------------------------------------------------
template <int KS>
__global__ void natten_kernel(const _Float16* __restrict__ qh, const _Float16* __restrict__ kh,
                              const _Float16* __restrict__ vh, const float* __restrict__ rpb,
                              _Float16* __restrict__ fused, int ch_off) {
  int pid = blockIdx.x * blockDim.x + threadIdx.x;
  if (pid >= NPIX) return;
  int x = pid % W_, y = (pid / W_) % H_, b = pid / HW_;
  const int R = KS / 2;
  int ni = y - R; if (ni < 0) ni = 0; if (ni > H_ - KS) ni = H_ - KS;
  int nj = x - R; if (nj < 0) nj = 0; if (nj > W_ - KS) nj = W_ - KS;
  int pi = KS - 1 + ni - y;
  int pj = KS - 1 + nj - x;

  float attn[KS * KS];
  #pragma unroll
  for (int p = 0; p < KS; ++p)
    #pragma unroll
    for (int q = 0; q < KS; ++q)
      attn[p * KS + q] = rpb[(pi + p) * (2 * KS - 1) + (pj + q)];

  const _Float16* qrow = qh + (size_t)pid * 64;
  for (int cb = 0; cb < 64; cb += 16) {
    float qv[16];
    #pragma unroll
    for (int t = 0; t < 16; ++t) qv[t] = (float)qrow[cb + t];
    #pragma unroll
    for (int p = 0; p < KS; ++p) {
      const _Float16* krow = kh + ((size_t)((b * H_ + ni + p) * W_ + nj)) * 64 + cb;
      #pragma unroll
      for (int q = 0; q < KS; ++q) {
        float s = 0.f;
        #pragma unroll
        for (int t = 0; t < 16; ++t) s += qv[t] * (float)krow[q * 64 + t];
        attn[p * KS + q] += s;
      }
    }
  }
  // softmax over KS*KS
  float mx = attn[0];
  #pragma unroll
  for (int i = 1; i < KS * KS; ++i) mx = fmaxf(mx, attn[i]);
  float sum = 0.f;
  #pragma unroll
  for (int i = 0; i < KS * KS; ++i) { attn[i] = __expf(attn[i] - mx); sum += attn[i]; }
  float inv = 1.0f / sum;
  #pragma unroll
  for (int i = 0; i < KS * KS; ++i) attn[i] *= inv;

  _Float16* orow = fused + (size_t)pid * 256 + ch_off;
  for (int cb = 0; cb < 64; cb += 16) {
    float acc[16];
    #pragma unroll
    for (int t = 0; t < 16; ++t) acc[t] = 0.f;
    #pragma unroll
    for (int p = 0; p < KS; ++p) {
      const _Float16* vrow = vh + ((size_t)((b * H_ + ni + p) * W_ + nj)) * 64 + cb;
      #pragma unroll
      for (int q = 0; q < KS; ++q) {
        float a = attn[p * KS + q];
        #pragma unroll
        for (int t = 0; t < 16; ++t) acc[t] += a * (float)vrow[q * 64 + t];
      }
    }
    #pragma unroll
    for (int t = 0; t < 16; ++t) orow[cb + t] = (_Float16)acc[t];
  }
}

// ---------------------------------------------------------------------------
// Stage 5: fusion 3x3 conv, im2col GEMM M=36864 K=9*256 N=64 (WMMA) + bias+relu.
// Block = 4 waves = one 16-pixel m-tile x all 64 output channels.
// The 3x18-pixel x 256ch f16 activation halo (27KB) is staged into LDS ONCE
// per block with GLOBAL_LOAD_ASYNC_TO_LDS_B128 (ASYNCcnt path); image-border
// halo entries are zero-filled with plain LDS stores (disjoint addresses).
// All 4 waves then run 72 LDS-fed WMMAs with no per-lane masking.
// ---------------------------------------------------------------------------
__global__ void gemm_fusion_kernel(const _Float16* __restrict__ fused, // [B,H,W,256]
                                   const _Float16* __restrict__ wt,    // [9][64][256]
                                   const float* __restrict__ bias,
                                   float* __restrict__ out) {          // [B,64,H,W]
  __shared__ _Float16 smem[54 * 256];   // [3 rows][18 cols][256 ch] = 27648 B
  int tidx = threadIdx.x;
  int lane = tidx & 31;
  int wave = tidx >> 5;
  int pid0 = blockIdx.x * 16;
  int x0 = pid0 % W_, y = (pid0 / W_) % H_, b = pid0 / HW_;

  // ---- stage A-halo into LDS: 54 slots x 512B, 16B per lane-task ----------
  for (int t = tidx; t < 54 * 32; t += 128) {
    int slot = t >> 5;          // 0..53 : row*18 + col
    int seg  = t & 31;          // 16B segment within the 512B slot
    int row = slot / 18, col = slot % 18;
    int yy = y + row - 1;
    int xx = x0 + col - 1;
    _Float16* lptr = &smem[slot * 256 + seg * 8];
    if (yy >= 0 && yy < H_ && xx >= 0 && xx < W_) {
      const _Float16* gptr = fused + ((size_t)((b * H_ + yy) * W_ + xx)) * 256 + seg * 8;
      unsigned lds_addr = (unsigned)(uintptr_t)lptr;   // low 32 bits = LDS offset
      unsigned long long gaddr = (unsigned long long)(uintptr_t)gptr;
      asm volatile("global_load_async_to_lds_b128 %0, %1, off"
                   :: "v"(lds_addr), "v"(gaddr) : "memory");
    } else {
      h8 z = {};
      *(h8*)lptr = z;                                   // zero the border halo
    }
  }
  asm volatile("s_wait_asynccnt 0" ::: "memory");
  __syncthreads();

  // ---- 72 LDS-fed WMMAs per wave -----------------------------------------
  int n0 = wave * 16;
  int sel = lane >> 4;
  int mr = lane & 15;
  int nc = lane & 15;
  v8f c = {};
  #pragma unroll
  for (int ky = 0; ky < 3; ++ky) {
    #pragma unroll
    for (int kx = 0; kx < 3; ++kx) {
      const _Float16* arow = &smem[(ky * 18 + mr + kx) * 256];
      const _Float16* brow = wt + ((size_t)((ky * 3 + kx) * 64 + n0 + nc)) * 256;
      #pragma unroll
      for (int kb = 0; kb < 256; kb += 32) {
        v16h a = load_a32(arow + kb, sel);
        v16h bm = load_b32(brow + kb, sel);
        c = __builtin_amdgcn_wmma_f32_16x16x32_f16(false, a, false, bm, (short)0, c, false, false);
      }
    }
  }
  float bz = bias[n0 + nc];
  #pragma unroll
  for (int r = 0; r < 8; ++r) {
    int m = r + 8 * sel;
    float v = fmaxf(c[r] + bz, 0.0f);
    out[((size_t)(b * C_ + n0 + nc) * H_ + y) * W_ + (x0 + m)] = v;
  }
}

// ---------------------------------------------------------------------------
// Host-side launcher.  Inputs in setup_inputs() dict order:
// 0 sup_frame, 1 key_frame, 2 conv_smooth_w, 3 conv_smooth_b, 4 conv_reduce_w,
// 5 conv_reduce_b, 6 v_w0, 7 v_b0, 8 rpb0, 9 v_w1, 10 v_b1, 11 rpb1,
// 12 v_w2, 13 v_b2, 14 rpb2, 15 fusion_w, 16 fusion_b
// ---------------------------------------------------------------------------
extern "C" void kernel_launch(void* const* d_in, const int* in_sizes, int n_in,
                              void* d_out, int out_size, void* d_ws, size_t ws_size,
                              hipStream_t stream) {
  const float* sup      = (const float*)d_in[0];
  const float* key      = (const float*)d_in[1];
  const float* w_smooth = (const float*)d_in[2];
  const float* b_smooth = (const float*)d_in[3];
  const float* w_reduce = (const float*)d_in[4];
  const float* b_reduce = (const float*)d_in[5];
  const float* v_w[3] = {(const float*)d_in[6], (const float*)d_in[9], (const float*)d_in[12]};
  const float* v_b[3] = {(const float*)d_in[7], (const float*)d_in[10], (const float*)d_in[13]};
  const float* rpb[3] = {(const float*)d_in[8], (const float*)d_in[11], (const float*)d_in[14]};
  const float* w_fus   = (const float*)d_in[15];
  const float* b_fus   = (const float*)d_in[16];
  float* out = (float*)d_out;

  // workspace layout (bytes), all 256B aligned
  char* ws = (char*)d_ws;
  size_t off = 0;
  _Float16* fused = (_Float16*)(ws + off); off += (size_t)NPIX * 256 * 2;  // 18.9 MB
  _Float16* x128  = (_Float16*)(ws + off); off += (size_t)NPIX * 128 * 2;  //  9.4 MB
  _Float16* sup2  = (_Float16*)(ws + off); off += (size_t)NPIX * 64 * 2;   //  4.7 MB
  _Float16* qh    = (_Float16*)(ws + off); off += (size_t)NPIX * 64 * 2;   //  4.7 MB
  _Float16* vh    = (_Float16*)(ws + off); off += (size_t)3 * NPIX * 64 * 2; // 14.2 MB
  _Float16* wr_h  = (_Float16*)(ws + off); off += 64 * 128 * 2;
  _Float16* vw_h  = (_Float16*)(ws + off); off += 3 * 64 * 64 * 2;
  _Float16* wft_h = (_Float16*)(ws + off); off += 9 * 64 * 256 * 2;
  (void)ws_size; // requires ~52.3 MB of workspace

  const int T = 256;
  // stage 0: layout prep + weight conversion
  prep_kernel<<<(NPIX * 64 + T - 1) / T, T, 0, stream>>>(sup, key, fused, x128, qh);
  cast_f2h<<<(64 * 128 + T - 1) / T, T, 0, stream>>>(w_reduce, wr_h, 64 * 128);
  for (int l = 0; l < 3; ++l)
    cast_f2h<<<(64 * 64 + T - 1) / T, T, 0, stream>>>(v_w[l], vw_h + l * 64 * 64, 64 * 64);
  wt_fusion_kernel<<<(64 * 256 * 9 + T - 1) / T, T, 0, stream>>>(w_fus, wft_h);

  // stage 1: shift + grouped smooth conv
  smooth_kernel<<<(NPIX * 64 + T - 1) / T, T, 0, stream>>>(sup, w_smooth, b_smooth, x128);

  // stage 2: conv_reduce (WMMA)
  gemm_reduce_kernel<<<NPIX / 16, 128, 0, stream>>>(x128, wr_h, b_reduce, sup2);

  // stage 3: v convs, all 3 levels (WMMA)
  gemm_v_kernel<<<dim3(NPIX / 16, 3), 128, 0, stream>>>(sup2, vw_h, v_b[0], v_b[1], v_b[2], vh);

  // stage 4: neighborhood attention levels -> fused channels 64/128/192
  natten_kernel<3><<<NPIX / 128, 128, 0, stream>>>(qh, sup2, vh,                 rpb[0], fused, 64);
  natten_kernel<5><<<NPIX / 128, 128, 0, stream>>>(qh, sup2, vh + (size_t)NPIX * 64,     rpb[1], fused, 128);
  natten_kernel<7><<<NPIX / 128, 128, 0, stream>>>(qh, sup2, vh + (size_t)2 * NPIX * 64, rpb[2], fused, 192);

  // stage 5: fusion conv (WMMA, LDS-staged A via async-to-LDS) -> d_out
  gemm_fusion_kernel<<<NPIX / 16, 128, 0, stream>>>(fused, wft_h, b_fus, out);
  (void)in_sizes; (void)n_in; (void)out_size;
}
